// FullAttention_8684423872677
// MI455X (gfx1250) — matrix operand order repaired and checked by reading, not compiled, not run on hardware
//
#include <hip/hip_runtime.h>
#include <hip/hip_bf16.h>

typedef __attribute__((ext_vector_type(16))) _Float16 v16h;
typedef __attribute__((ext_vector_type(8)))  _Float16 v8h;
typedef __attribute__((ext_vector_type(4)))  _Float16 v4h;
typedef __attribute__((ext_vector_type(8)))  float    v8f;
typedef __attribute__((ext_vector_type(4)))  unsigned int u32x4;
typedef __attribute__((ext_vector_type(8)))  int      i32x8;
typedef __attribute__((ext_vector_type(4)))  int      i32x4;

#define S_LEN 2048
#define D_DIM 2048
#define NQH   32
#define NKVH  8
#define HD    64
#define OPSZ  3072   // 32*64 + 2*8*64

// Detect the TDM builtin (tensor data mover).  Arity differs per toolchain:
// ROCm 7.2 (clang-22): 5 args; amdgpu-toolchain (clang-23): 6 args.
#if defined(__has_builtin)
#if __has_builtin(__builtin_amdgcn_tensor_load_to_lds) && \
    __has_builtin(__builtin_amdgcn_s_wait_tensorcnt)
#define USE_TDM 1
#endif
#endif
#ifndef USE_TDM
#define USE_TDM 0
#endif

// ---------------------------------------------------------------------------
// helpers
// ---------------------------------------------------------------------------
__device__ __forceinline__ v16h combine8(v8h lo, v8h hi) {
  v16h r;
#pragma unroll
  for (int i = 0; i < 8; ++i) { r[i] = lo[i]; r[i + 8] = hi[i]; }
  return r;
}

__device__ __forceinline__ v8f wmma_f16(v16h a, v16h b, v8f c) {
  return __builtin_amdgcn_wmma_f32_16x16x32_f16(false, a, false, b,
                                                (short)0, c, false, false);
}

// DPP16 cross-lane ops: butterfly over the 16-lane row (both halves of the
// wave independently) -- pure VALU, no LDS traffic.
template <int CTRL>
__device__ __forceinline__ float dppf(float x) {
  return __int_as_float(__builtin_amdgcn_update_dpp(
      0, __float_as_int(x), CTRL, 0xF, 0xF, true));
}
__device__ __forceinline__ float row_max16(float x) {
  x = fmaxf(x, dppf<0xB1>(x));   // xor 1 (quad_perm 1,0,3,2)
  x = fmaxf(x, dppf<0x4E>(x));   // xor 2 (quad_perm 2,3,0,1)
  x = fmaxf(x, dppf<0x141>(x));  // xor 4 (row_half_mirror)
  x = fmaxf(x, dppf<0x140>(x));  // xor 8 (row_mirror)
  return x;
}
__device__ __forceinline__ float row_sum16(float x) {
  x += dppf<0xB1>(x);
  x += dppf<0x4E>(x);
  x += dppf<0x141>(x);
  x += dppf<0x140>(x);
  return x;
}

__device__ __forceinline__ unsigned lds_addr(const void* p) {
  return (unsigned)(unsigned long long)(uintptr_t)p;  // low 32 bits = LDS offset
}

#if USE_TDM
// Issue a TDM 2D tile load: tile_h rows of tile_w f16 elements, global row
// stride `stride_elems`, LDS rows padded by pad_amt_code+1 DWORDs every
// 2^(pad_int_code+1) DWORDs (D# group1 pad fields).
__device__ __forceinline__ void tdm_load_2d(unsigned lds, const void* gptr,
                                            unsigned tile_w, unsigned tile_h,
                                            unsigned stride_elems,
                                            unsigned pad_int_code,
                                            unsigned pad_amt_code) {
  unsigned long long ga = (unsigned long long)(uintptr_t)gptr;
  u32x4 g0;
  g0[0] = 1u;                                   // count=1, user descriptor
  g0[1] = lds;                                  // lds_addr
  g0[2] = (unsigned)ga;                         // global_addr[95:64]
  g0[3] = (unsigned)((ga >> 32) & 0x1FFFFFFu) | (2u << 30);  // addr hi | type=2
  i32x8 g1;
  g1[0] = (int)((1u << 16) |                    // data_size = 2 bytes
                (1u << 20) |                    // pad_enable
                (pad_int_code << 22) | (pad_amt_code << 25));
  g1[1] = (int)((tile_w & 0xFFFFu) << 16);      // tensor_dim0[15:0]
  g1[2] = (int)((tile_w >> 16) | ((tile_h & 0xFFFFu) << 16));  // td0 hi | td1 lo
  g1[3] = (int)((tile_h >> 16) | (tile_w << 16));              // td1 hi | tile_dim0
  g1[4] = (int)tile_h;                          // tile_dim1 | tile_dim2=0
  g1[5] = (int)stride_elems;                    // tensor_dim0_stride[31:0]
  g1[6] = 0;
  g1[7] = 0;
  i32x4 g2 = {0, 0, 0, 0};
  i32x4 g3 = {0, 0, 0, 0};
#if __clang_major__ >= 23
  i32x8 g4 = {0, 0, 0, 0, 0, 0, 0, 0};
  __builtin_amdgcn_tensor_load_to_lds(g0, g1, g2, g3, g4, 0);
#else
  __builtin_amdgcn_tensor_load_to_lds(g0, g1, g2, g3, 0);
#endif
}
#endif

// ---------------------------------------------------------------------------
// prep kernels: one-time conversions so every hot-loop tile is contiguous f16
// ---------------------------------------------------------------------------
__global__ __launch_bounds__(256)
void cast_f16x4(const float* __restrict__ in, _Float16* __restrict__ out) {
  size_t i = ((size_t)blockIdx.x * 256 + threadIdx.x) * 4;
  float4 f = *(const float4*)(in + i);
  v4h h = {(_Float16)f.x, (_Float16)f.y, (_Float16)f.z, (_Float16)f.w};
  *(v4h*)(out + i) = h;
}

// W[R][Cc] f32 -> Wt[Cc][R] f16 (N-major weights for K-contiguous tiles)
__global__ __launch_bounds__(256)
void transpose_cast(const float* __restrict__ W, _Float16* __restrict__ Wt,
                    int R, int Cc) {
  size_t gid = (size_t)blockIdx.x * 256 + threadIdx.x;
  int r = (int)(gid / Cc), c = (int)(gid % Cc);
  Wt[(size_t)c * R + r] = (_Float16)W[gid];
}

// ---------------------------------------------------------------------------
// GEMM: C[M,N](f32) = A[M,K](f16) * Bt[N,K](f16, N-major).
// 64x128 block tile, BK=32, 4 waves x (32x64), TDM double-buffered staging.
// ---------------------------------------------------------------------------
#define BK  32
#define LDP 40   // padded LDS row stride in halves (conflict-free, 16B aligned)

__global__ __launch_bounds__(128)
void gemm_f16(const _Float16* __restrict__ A, const _Float16* __restrict__ Bt,
              float* __restrict__ C, int M, int N, int K) {
  __shared__ _Float16 As[2][64 * LDP];
  __shared__ _Float16 Bs[2][128 * LDP];

  const int t    = threadIdx.x;
  const int lane = t & 31, wave = t >> 5;
  const int g    = lane >> 4, ln = lane & 15;
  const int m0   = blockIdx.y * 64;
  const int n0   = blockIdx.x * 128;
  const int wm   = (wave >> 1) * 32;
  const int wn   = (wave & 1) * 64;

  v8f acc[2][4] = {};
  int buf = 0;

#if USE_TDM
  if (wave == 0) {
    tdm_load_2d(lds_addr(&As[0][0]), A + (size_t)m0 * K, BK, 64, K, 3, 3);
    tdm_load_2d(lds_addr(&Bs[0][0]), Bt + (size_t)n0 * K, BK, 128, K, 3, 3);
  }
#endif

  for (int k0 = 0; k0 < K; k0 += BK) {
#if USE_TDM
    if (wave == 0) __builtin_amdgcn_s_wait_tensorcnt(0);
    __syncthreads();                      // current buffer ready for all waves
    if (k0 + BK < K && wave == 0) {       // prefetch next block into other buf
      tdm_load_2d(lds_addr(&As[buf ^ 1][0]), A + (size_t)m0 * K + k0 + BK,
                  BK, 64, K, 3, 3);
      tdm_load_2d(lds_addr(&Bs[buf ^ 1][0]), Bt + (size_t)n0 * K + k0 + BK,
                  BK, 128, K, 3, 3);
    }
#else
    __syncthreads();
#pragma unroll
    for (int j = 0; j < 2; ++j) {         // A: 64x32 = 256 v8h chunks
      int idx = t + j * 128, row = idx >> 2, q8 = idx & 3;
      *(v8h*)&As[0][row * LDP + q8 * 8] =
          *(const v8h*)(A + (size_t)(m0 + row) * K + k0 + q8 * 8);
    }
#pragma unroll
    for (int j = 0; j < 4; ++j) {         // B: 128x32 = 512 v8h chunks
      int idx = t + j * 128, row = idx >> 2, q8 = idx & 3;
      *(v8h*)&Bs[0][row * LDP + q8 * 8] =
          *(const v8h*)(Bt + (size_t)(n0 + row) * K + k0 + q8 * 8);
    }
    __syncthreads();
#endif
    const _Float16* as = &As[buf][0];
    const _Float16* bs = &Bs[buf][0];
    v16h a[2], b[4];
#pragma unroll
    for (int mt = 0; mt < 2; ++mt) {
      const _Float16* p = as + (wm + mt * 16 + ln) * LDP;
      a[mt] = combine8(*(const v8h*)(p + 8 * g), *(const v8h*)(p + 16 + 8 * g));
    }
#pragma unroll
    for (int nt = 0; nt < 4; ++nt) {
      const _Float16* p = bs + (wn + nt * 16 + ln) * LDP;
      b[nt] = combine8(*(const v8h*)(p + 8 * g), *(const v8h*)(p + 16 + 8 * g));
    }
#pragma unroll
    for (int mt = 0; mt < 2; ++mt)
#pragma unroll
      for (int nt = 0; nt < 4; ++nt)
        acc[mt][nt] = wmma_f16(a[mt], b[nt], acc[mt][nt]);
#if USE_TDM
    buf ^= 1;
#endif
  }

#pragma unroll
  for (int mt = 0; mt < 2; ++mt)
#pragma unroll
    for (int nt = 0; nt < 4; ++nt)
#pragma unroll
      for (int r = 0; r < 8; ++r) {
        int m = m0 + wm + mt * 16 + r + 8 * g;
        int n = n0 + wn + nt * 16 + ln;
        C[(size_t)m * N + n] = acc[mt][nt][r];
      }
}

// ---------------------------------------------------------------------------
// Rotary + split: qkv f32 [S][3072] -> Qh[H][S][64], Kh[KV][S][64] (f16),
// V written TRANSPOSED: Vt[KV][HD][S] so PV B-fragments need no LDS transpose.
// ---------------------------------------------------------------------------
__global__ __launch_bounds__(256)
void rope_split(const float* __restrict__ qkv, _Float16* __restrict__ Qh,
                _Float16* __restrict__ Kh, _Float16* __restrict__ Vt) {
  const int gid = blockIdx.x * 256 + threadIdx.x;
  const int s = gid / OPSZ;
  const int c = gid - s * OPSZ;
  const float* row = qkv + (size_t)s * OPSZ;

  if (c < NQH * HD) {                       // Q
    const int h = c >> 6, hd = c & 63;
    const float* x = row + h * HD;
    float v;
    if (hd < 32) {
      float f  = (float)s * __powf(100000.0f, -(float)hd * (1.0f / 32.0f));
      float cs = __cosf(f), sn = __sinf(f);
      v = (hd < 16) ? (x[hd] * cs - x[hd + 16] * sn)
                    : (x[hd] * cs + x[hd - 16] * sn);
    } else {
      v = x[hd];
    }
    Qh[((size_t)h * S_LEN + s) * HD + hd] = (_Float16)v;
  } else if (c < NQH * HD + NKVH * HD) {    // K
    const int cc = c - NQH * HD;
    const int h = cc >> 6, hd = cc & 63;
    const float* x = row + NQH * HD + h * HD;
    float v;
    if (hd < 32) {
      float f  = (float)s * __powf(100000.0f, -(float)hd * (1.0f / 32.0f));
      float cs = __cosf(f), sn = __sinf(f);
      v = (hd < 16) ? (x[hd] * cs - x[hd + 16] * sn)
                    : (x[hd] * cs + x[hd - 16] * sn);
    } else {
      v = x[hd];
    }
    Kh[((size_t)h * S_LEN + s) * HD + hd] = (_Float16)v;
  } else {                                  // V (transposed store)
    const int cc = c - NQH * HD - NKVH * HD;
    const int h = cc >> 6, hd = cc & 63;
    Vt[((size_t)h * HD + hd) * S_LEN + s] = (_Float16)row[c];
  }
}

// ---------------------------------------------------------------------------
// Flash attention (causal, GQA=4). grid=(S/64, H), 128 threads (4 waves),
// wave owns 16 q-rows. K/V blocks of 32 keys TDM-double-buffered in LDS,
// DPP16 online softmax, f16 WMMA with f32 accumulation.
// ---------------------------------------------------------------------------
#define KSP 72   // Ks LDS row stride (64 + 8 pad halves)
#define VSP 40   // Vs LDS row stride (32 + 8 pad halves)
#define PSP 40   // Ps LDS row stride

__global__ __launch_bounds__(128)
void attn(const _Float16* __restrict__ Qh, const _Float16* __restrict__ Kh,
          const _Float16* __restrict__ Vt, _Float16* __restrict__ aout) {
  __shared__ _Float16 Ks[2][32 * KSP];     // [key][hd], padded
  __shared__ _Float16 Vs[2][64 * VSP];     // [hd][key], padded (V pre-transposed)
  __shared__ _Float16 Ps[4][16 * PSP];     // per-wave P tile

  const int qb   = blockIdx.x;
  const int h    = blockIdx.y;
  const int kv   = h >> 2;
  const int t    = threadIdx.x;
  const int lane = t & 31, wave = t >> 5;
  const int g    = lane >> 4, ln = lane & 15;
  const int q0   = wave * 16;

  const _Float16* Q  = Qh + ((size_t)h * S_LEN + qb * 64) * HD;
  const _Float16* Kb = Kh + (size_t)kv * S_LEN * HD;
  const _Float16* Vb = Vt + (size_t)kv * HD * S_LEN;

  v16h qf[2];
#pragma unroll
  for (int f = 0; f < 2; ++f) {
    const _Float16* qr = Q + (q0 + ln) * HD + f * 32;
    qf[f] = combine8(*(const v8h*)(qr + 8 * g), *(const v8h*)(qr + 16 + 8 * g));
  }

  float mrow[8], lrow[8];
#pragma unroll
  for (int r = 0; r < 8; ++r) { mrow[r] = -INFINITY; lrow[r] = 0.0f; }
  v8f o[4] = {};

  const int nkb = qb * 2 + 2;
  int buf = 0;

#if USE_TDM
  if (wave == 0) {
    tdm_load_2d(lds_addr(&Ks[0][0]), Kb, HD, 32, HD, 4, 3);
    tdm_load_2d(lds_addr(&Vs[0][0]), Vb, 32, HD, S_LEN, 3, 3);
  }
#endif

  for (int kb = 0; kb < nkb; ++kb) {
#if USE_TDM
    if (wave == 0) __builtin_amdgcn_s_wait_tensorcnt(0);
    __syncthreads();
    if (kb + 1 < nkb && wave == 0) {
      tdm_load_2d(lds_addr(&Ks[buf ^ 1][0]),
                  Kb + (size_t)(kb + 1) * 32 * HD, HD, 32, HD, 4, 3);
      tdm_load_2d(lds_addr(&Vs[buf ^ 1][0]),
                  Vb + (size_t)(kb + 1) * 32, 32, HD, S_LEN, 3, 3);
    }
#else
    __syncthreads();
#pragma unroll
    for (int j = 0; j < 2; ++j) {           // K: 32x64 = 256 v8h chunks
      int idx = t + j * 128, row = idx >> 3, q8 = idx & 7;
      *(v8h*)&Ks[0][row * KSP + q8 * 8] =
          *(const v8h*)(Kb + (size_t)(kb * 32 + row) * HD + q8 * 8);
    }
#pragma unroll
    for (int j = 0; j < 2; ++j) {           // V: 64x32 = 256 v8h chunks
      int idx = t + j * 128, row = idx >> 2, q8 = idx & 3;
      *(v8h*)&Vs[0][row * VSP + q8 * 8] =
          *(const v8h*)(Vb + (size_t)row * S_LEN + kb * 32 + q8 * 8);
    }
    __syncthreads();
#endif
    const _Float16* ks = &Ks[buf][0];
    const _Float16* vs = &Vs[buf][0];

    // scores: 16 rows x 32 keys per wave
    v8f sc[2] = {};
#pragma unroll
    for (int nt = 0; nt < 2; ++nt) {
#pragma unroll
      for (int f = 0; f < 2; ++f) {
        const _Float16* kr = ks + (nt * 16 + ln) * KSP + f * 32;
        v16h bf = combine8(*(const v8h*)(kr + 8 * g),
                           *(const v8h*)(kr + 16 + 8 * g));
        sc[nt] = wmma_f16(qf[f], bf, sc[nt]);
      }
    }

    // mask + online softmax (DPP16 row reductions)
    const float scale = 0.125f;
#pragma unroll
    for (int r = 0; r < 8; ++r) {
      const int qi  = qb * 64 + q0 + r + 8 * g;
      const int k0i = kb * 32 + ln;
      float s0 = sc[0][r] * scale + ((k0i > qi) ? -1e9f : 0.0f);
      float s1 = sc[1][r] * scale + ((k0i + 16 > qi) ? -1e9f : 0.0f);
      const float mnew  = fmaxf(mrow[r], row_max16(fmaxf(s0, s1)));
      const float alpha = __expf(mrow[r] - mnew);
      const float p0 = __expf(s0 - mnew);
      const float p1 = __expf(s1 - mnew);
      lrow[r] = lrow[r] * alpha + row_sum16(p0 + p1);
      mrow[r] = mnew;
#pragma unroll
      for (int nt = 0; nt < 4; ++nt) o[nt][r] *= alpha;
      Ps[wave][(r + 8 * g) * PSP + ln]      = (_Float16)p0;
      Ps[wave][(r + 8 * g) * PSP + 16 + ln] = (_Float16)p1;
    }

    // P fragment (C-layout -> A-layout via per-wave LDS)
    const _Float16* pr = &Ps[wave][ln * PSP];
    v16h pf = combine8(*(const v8h*)(pr + 8 * g), *(const v8h*)(pr + 16 + 8 * g));

    // O += P @ V
#pragma unroll
    for (int nt = 0; nt < 4; ++nt) {
      const _Float16* vr = vs + (nt * 16 + ln) * VSP;
      v16h vf = combine8(*(const v8h*)(vr + 8 * g),
                         *(const v8h*)(vr + 16 + 8 * g));
      o[nt] = wmma_f16(pf, vf, o[nt]);
    }
#if USE_TDM
    buf ^= 1;
#endif
  }

#pragma unroll
  for (int r = 0; r < 8; ++r) {
    const float inv = 1.0f / lrow[r];
    const int qi = qb * 64 + q0 + r + 8 * g;
#pragma unroll
    for (int nt = 0; nt < 4; ++nt)
      aout[(size_t)qi * (NQH * HD) + h * HD + nt * 16 + ln] =
          (_Float16)(o[nt][r] * inv);
  }
}

// ---------------------------------------------------------------------------
// launch
// ---------------------------------------------------------------------------
extern "C" void kernel_launch(void* const* d_in, const int* in_sizes, int n_in,
                              void* d_out, int out_size, void* d_ws,
                              size_t ws_size, hipStream_t stream) {
  const float* X    = (const float*)d_in[0];
  const float* Wqkv = (const float*)d_in[2];
  const float* Wo   = (const float*)d_in[3];
  float* out        = (float*)d_out;

  char* ws = (char*)d_ws;
  size_t off = 0;
  auto alloc = [&](size_t bytes) { char* p = ws + off; off += (bytes + 255) & ~255ull; return p; };
  _Float16* Xh    = (_Float16*)alloc((size_t)S_LEN * D_DIM * 2);
  _Float16* WqkvT = (_Float16*)alloc((size_t)OPSZ * D_DIM * 2);
  _Float16* WoT   = (_Float16*)alloc((size_t)D_DIM * D_DIM * 2);
  float*    qkv   = (float*)   alloc((size_t)S_LEN * OPSZ * 4);
  _Float16* Qh    = (_Float16*)alloc((size_t)NQH * S_LEN * HD * 2);
  _Float16* Kh    = (_Float16*)alloc((size_t)NKVH * S_LEN * HD * 2);
  _Float16* Vt    = (_Float16*)alloc((size_t)NKVH * S_LEN * HD * 2);
  _Float16* aout  = (_Float16*)alloc((size_t)S_LEN * D_DIM * 2);

  // prep: one-time conversions to f16 (weights transposed to N-major)
  cast_f16x4<<<(S_LEN * D_DIM) / 1024, 256, 0, stream>>>(X, Xh);
  transpose_cast<<<(D_DIM * OPSZ) / 256, 256, 0, stream>>>(Wqkv, WqkvT, D_DIM, OPSZ);
  transpose_cast<<<(D_DIM * D_DIM) / 256, 256, 0, stream>>>(Wo, WoT, D_DIM, D_DIM);

  // 1) QKV projection
  gemm_f16<<<dim3(OPSZ / 128, S_LEN / 64), 128, 0, stream>>>(
      Xh, WqkvT, qkv, S_LEN, OPSZ, D_DIM);

  // 2) rotary + head split (V transposed)
  rope_split<<<(S_LEN * OPSZ) / 256, 256, 0, stream>>>(qkv, Qh, Kh, Vt);

  // 3) causal flash attention
  attn<<<dim3(S_LEN / 64, NQH), 128, 0, stream>>>(Qh, Kh, Vt, aout);

  // 4) output projection
  gemm_f16<<<dim3(D_DIM / 128, S_LEN / 64), 128, 0, stream>>>(
      aout, WoT, out, S_LEN, D_DIM, NQH * HD);
}
